// Parser_56470230008204
// MI455X (gfx1250) — matrix operand order repaired and checked by reading, not compiled, not run on hardware
//
#include <hip/hip_runtime.h>
#include <hip/hip_bf16.h>

// ---------------------------------------------------------------------------
// Problem constants (match reference: B=64, S=32, H=1024, R=256, DEPTH=9)
// ---------------------------------------------------------------------------
#define TB 64
#define TS 32
#define TH 1024
#define TR 256
#define TDEPTH 9

typedef __attribute__((ext_vector_type(16))) __bf16 v16bf;
typedef __attribute__((ext_vector_type(8)))  float  v8f;

// Tensor Data Mover availability (this toolchain: clang-23 / 6-arg builtin)
#if defined(__has_builtin)
#  if __has_builtin(__builtin_amdgcn_tensor_load_to_lds) && \
      __has_builtin(__builtin_amdgcn_s_wait_tensorcnt)
#    define USE_TDM 1
#  endif
#endif
#ifndef USE_TDM
#  define USE_TDM 0
#endif

#if USE_TDM
typedef __attribute__((ext_vector_type(4))) unsigned int v4u;
typedef __attribute__((ext_vector_type(8))) int          v8i;
typedef __attribute__((ext_vector_type(4))) int          v4i;
#endif

// ---------------------------------------------------------------------------
// Scalar helpers
// ---------------------------------------------------------------------------
__device__ __forceinline__ float bf2f(unsigned short h) {
  return __uint_as_float(((unsigned int)h) << 16);
}
__device__ __forceinline__ unsigned short f2bf(float f) {
  unsigned int u = __float_as_uint(f);
  unsigned int r = u + 0x7FFFu + ((u >> 16) & 1u);   // round-to-nearest-even
  return (unsigned short)(r >> 16);
}
__device__ __forceinline__ float selu_f(float x) {
  const float scale = 1.0507009873554805f;
  const float alpha = 1.6732632423543772f;
  return x > 0.0f ? scale * x : scale * alpha * (__expf(x) - 1.0f);
}
__device__ __forceinline__ float sigmoid_f(float x) {
  return 1.0f / (1.0f + __expf(-x));
}

// ---------------------------------------------------------------------------
// Elementwise cast kernels
// ---------------------------------------------------------------------------
__global__ void cast_bf16_k(const float* __restrict__ in,
                            unsigned short* __restrict__ out, int n) {
  int i = blockIdx.x * blockDim.x + threadIdx.x;
  if (i < n) out[i] = f2bf(in[i]);
}
__global__ void cast_selu_bf16_k(const float* __restrict__ in,
                                 unsigned short* __restrict__ out, int n) {
  int i = blockIdx.x * blockDim.x + threadIdx.x;
  if (i < n) out[i] = f2bf(selu_f(in[i]));
}

// Root mean over sequence: xc[b,h] = mean_s y[b*S+s, h]  (f32 in, bf16 out)
__global__ void mean_rows_k(const float* __restrict__ y,
                            unsigned short* __restrict__ xc) {
  int i = blockIdx.x * blockDim.x + threadIdx.x;
  if (i >= TB * TH) return;
  int b = i >> 10, h = i & (TH - 1);
  const float* p = y + (size_t)b * TS * TH + h;
  float s = 0.0f;
#pragma unroll
  for (int t = 0; t < TS; ++t) s += p[(size_t)t * TH];
  xc[i] = f2bf(s * (1.0f / TS));
}

// ---------------------------------------------------------------------------
// WMMA bf16 GEMM:  C[M,N] = sum_segs A_s[M,K_s] * W_s[N,K_s]^T  + bias
// Block: 256 threads = 8 waves; wave tile 16(M) x 64(N); block tile 64x128.
// 16-bit fragment layouts used here (A and B share one k order):
//   A (16x32, MxK):  lane -> row (lane&15); khalf=lane>>4 selects K phase.
//                    halves 0..15 = A[m, k + 16*khalf .. +15]  (contiguous)
//   B (32x16, KxN):  lane -> column n (lane&15) == row of W (C = A*W^T).
//                    halves 0..15 = W[n, k + 16*khalf .. +15]  (contiguous)
//   C (16x16 f32):   vgpr r, lane l -> (m = r + 8*(l>>4), n = l&15)
// ---------------------------------------------------------------------------
struct GemmSeg {
  const unsigned short* A; int lda; int K;
  const unsigned short* W; int ldw;
};
struct GemmArgs {
  GemmSeg s0, s1;                 // s1.K == 0 -> unused
  const float* bias;
  float*          outF; int ldo;  // EPI 0/1
  unsigned short* outB;           // EPI 2/3
  const unsigned short* resid;    // EPI 3: parent xc (bf16, ld 1024)
  int rowAdd;                     // EPI 3: child bit (out row = 2m + rowAdd)
};

union Frag { uint4 u[2]; v16bf v; };

#if USE_TDM
// LDS staging geometry: row of 32 bf16 = 64B, TDM pads +16B -> stride 80B
#define SROW 40                   // LDS row stride in halves (80 bytes)
#define A_BUF (64 * SROW)         // one A stage (64 rows)
#define B_BUF (128 * SROW)        // one B stage (128 rows)

// Issue one 2D TDM tile load: rows x 32 bf16 elements, row stride lda
// (elements), into LDS at ldsOff with 16B/row padding (bank-conflict-free).
__device__ __forceinline__ void tdm_issue(unsigned ldsOff,
                                          const unsigned short* gp,
                                          int rows, int lda) {
  unsigned long long ga = (unsigned long long)(uintptr_t)gp;
  v4u g0 = { 1u,                                    // count=1 (user descriptor)
             ldsOff,                                // lds_addr
             (unsigned)ga,                          // global_addr[31:0]
             (unsigned)((ga >> 32) & 0x1FFFFFFu) | (2u << 30) };  // type=2
  unsigned td0 = (unsigned)lda;                     // tensor_dim0 = row length
  v8i g1;
  // data_size=2B(code1) | pad_enable | pad_interval=16dw(code3) |
  // pad_amount=4dw(code3)
  g1[0] = (int)((1u << 16) | (1u << 20) | (3u << 22) | (3u << 25));
  g1[1] = (int)((td0 & 0xFFFFu) << 16);             // tensor_dim0[15:0]
  g1[2] = (int)((td0 >> 16) & 0xFFFFu);             // td0[31:16] | td1[15:0]=0
  g1[3] = (int)(16u | (32u << 16));                 // td1=1<<20 | tile_dim0=32
  g1[4] = (int)(rows & 0xFFFF);                     // tile_dim1 | tile_dim2=0
  g1[5] = (int)td0;                                 // dim0_stride[31:0]
  g1[6] = 0;                                        // stride hi | dim1_stride
  g1[7] = 0;
  v4i zz4 = { 0, 0, 0, 0 };
  v8i zz8 = { 0, 0, 0, 0, 0, 0, 0, 0 };
  __builtin_amdgcn_tensor_load_to_lds(g0, g1, zz4, zz4, zz8, 0);
}

// Double-buffered TDM pipeline: wave 0 DMAs the next K-step's A/B tiles into
// the alternate LDS stage while all 8 waves run WMMAs out of the current one.
__device__ __forceinline__ void gemm_accum_tdm(
    v8f& c0, v8f& c1, v8f& c2, v8f& c3,
    unsigned short* sA, unsigned short* sB,       // [2][A_BUF] / [2][B_BUF]
    const unsigned short* __restrict__ A, int lda, int K,
    const unsigned short* __restrict__ W, int ldw,
    int mblk, int nblk, int wave, int wm, int wn, int lrow, int khalf) {
  const unsigned aOff0 = (unsigned)(uintptr_t)sA;
  const unsigned bOff0 = (unsigned)(uintptr_t)sB;
  const unsigned aOff[2] = { aOff0, aOff0 + A_BUF * 2u };
  const unsigned bOff[2] = { bOff0, bOff0 + B_BUF * 2u };
  const unsigned short* aBase = A + (size_t)mblk * lda;
  const unsigned short* wBase = W + (size_t)nblk * ldw;

  if (wave == 0) {
    tdm_issue(aOff[0], aBase, 64, lda);
    tdm_issue(bOff[0], wBase, 128, ldw);
    __builtin_amdgcn_s_wait_tensorcnt((short)0);
  }
  __syncthreads();

  const int aFragOff = (wm * 16 + lrow) * SROW + khalf * 16;
  const int bFragOff = (wn * 64 + lrow) * SROW + khalf * 16;
  const int nsteps = K >> 5;
  for (int s = 0; s < nsteps; ++s) {
    const int cur = s & 1, nxt = cur ^ 1;
    if (s + 1 < nsteps && wave == 0) {        // DMA next stage while computing
      const int k = (s + 1) << 5;
      tdm_issue(aOff[nxt], aBase + k, 64, lda);
      tdm_issue(bOff[nxt], wBase + k, 128, ldw);
    }
    Frag a, b0, b1, b2, b3;
    const unsigned short* pa  = sA + cur * A_BUF + aFragOff;
    const unsigned short* pb  = sB + cur * B_BUF + bFragOff;
    a.u[0]  = *(const uint4*)pa;
    a.u[1]  = *(const uint4*)(pa + 8);
    b0.u[0] = *(const uint4*)(pb + 0 * 16 * SROW);
    b0.u[1] = *(const uint4*)(pb + 0 * 16 * SROW + 8);
    b1.u[0] = *(const uint4*)(pb + 1 * 16 * SROW);
    b1.u[1] = *(const uint4*)(pb + 1 * 16 * SROW + 8);
    b2.u[0] = *(const uint4*)(pb + 2 * 16 * SROW);
    b2.u[1] = *(const uint4*)(pb + 2 * 16 * SROW + 8);
    b3.u[0] = *(const uint4*)(pb + 3 * 16 * SROW);
    b3.u[1] = *(const uint4*)(pb + 3 * 16 * SROW + 8);
    c0 = __builtin_amdgcn_wmma_f32_16x16x32_bf16(false, a.v, false, b0.v, (short)0, c0, false, false);
    c1 = __builtin_amdgcn_wmma_f32_16x16x32_bf16(false, a.v, false, b1.v, (short)0, c1, false, false);
    c2 = __builtin_amdgcn_wmma_f32_16x16x32_bf16(false, a.v, false, b2.v, (short)0, c2, false, false);
    c3 = __builtin_amdgcn_wmma_f32_16x16x32_bf16(false, a.v, false, b3.v, (short)0, c3, false, false);
    if (s + 1 < nsteps) {
      if (wave == 0) __builtin_amdgcn_s_wait_tensorcnt((short)0);
      __syncthreads();                        // stage handoff
    }
  }
  __syncthreads();                            // segment boundary safety
}
#endif  // USE_TDM

// Fallback: fragments straight from global (round-1 path).
__device__ __forceinline__ void gemm_accum_global(
    v8f& c0, v8f& c1, v8f& c2, v8f& c3,
    const unsigned short* __restrict__ A, int lda, int K,
    const unsigned short* __restrict__ W, int ldw,
    int m0, int n0, int lrow, int khalf) {
  const unsigned short* arow = A + (size_t)(m0 + lrow) * lda + khalf * 16;
  const unsigned short* w0 = W + (size_t)(n0 +  0 + lrow) * ldw + khalf * 16;
  const unsigned short* w1 = W + (size_t)(n0 + 16 + lrow) * ldw + khalf * 16;
  const unsigned short* w2 = W + (size_t)(n0 + 32 + lrow) * ldw + khalf * 16;
  const unsigned short* w3 = W + (size_t)(n0 + 48 + lrow) * ldw + khalf * 16;
  for (int k = 0; k < K; k += 32) {
    if (k + 32 < K) {
      __builtin_prefetch(arow + k + 32, 0, 3);
      __builtin_prefetch(w0 + k + 32, 0, 3);
    }
    Frag a, b0, b1, b2, b3;
    a.u[0]  = *(const uint4*)(arow + k);
    a.u[1]  = *(const uint4*)(arow + k + 8);
    b0.u[0] = *(const uint4*)(w0 + k);  b0.u[1] = *(const uint4*)(w0 + k + 8);
    b1.u[0] = *(const uint4*)(w1 + k);  b1.u[1] = *(const uint4*)(w1 + k + 8);
    b2.u[0] = *(const uint4*)(w2 + k);  b2.u[1] = *(const uint4*)(w2 + k + 8);
    b3.u[0] = *(const uint4*)(w3 + k);  b3.u[1] = *(const uint4*)(w3 + k + 8);
    c0 = __builtin_amdgcn_wmma_f32_16x16x32_bf16(false, a.v, false, b0.v, (short)0, c0, false, false);
    c1 = __builtin_amdgcn_wmma_f32_16x16x32_bf16(false, a.v, false, b1.v, (short)0, c1, false, false);
    c2 = __builtin_amdgcn_wmma_f32_16x16x32_bf16(false, a.v, false, b2.v, (short)0, c2, false, false);
    c3 = __builtin_amdgcn_wmma_f32_16x16x32_bf16(false, a.v, false, b3.v, (short)0, c3, false, false);
  }
}

// EPI: 0 = f32 store (+bias); 1 = selu -> f32; 2 = selu -> bf16;
//      3 = bf16 store of (resid - selu(acc+bias)) at row 2m+rowAdd (ld 1024)
template <int EPI>
__global__ __launch_bounds__(256)
void gemm_bf16_wmma(GemmArgs g) {
  const int lane  = threadIdx.x & 31;
  const int wave  = threadIdx.x >> 5;
  const int wm    = wave & 3, wn = wave >> 2;
  const int m0    = blockIdx.y * 64 + wm * 16;
  const int n0    = blockIdx.x * 128 + wn * 64;
  const int lrow  = lane & 15;
  const int khalf = lane >> 4;

  v8f c0 = {}, c1 = {}, c2 = {}, c3 = {};
#if USE_TDM
  __shared__ __align__(16) unsigned short sA[2 * A_BUF];
  __shared__ __align__(16) unsigned short sB[2 * B_BUF];
  gemm_accum_tdm(c0, c1, c2, c3, sA, sB, g.s0.A, g.s0.lda, g.s0.K,
                 g.s0.W, g.s0.ldw, blockIdx.y * 64, blockIdx.x * 128,
                 wave, wm, wn, lrow, khalf);
  if (g.s1.K > 0)
    gemm_accum_tdm(c0, c1, c2, c3, sA, sB, g.s1.A, g.s1.lda, g.s1.K,
                   g.s1.W, g.s1.ldw, blockIdx.y * 64, blockIdx.x * 128,
                   wave, wm, wn, lrow, khalf);
#else
  gemm_accum_global(c0, c1, c2, c3, g.s0.A, g.s0.lda, g.s0.K, g.s0.W,
                    g.s0.ldw, m0, n0, lrow, khalf);
  if (g.s1.K > 0)
    gemm_accum_global(c0, c1, c2, c3, g.s1.A, g.s1.lda, g.s1.K, g.s1.W,
                      g.s1.ldw, m0, n0, lrow, khalf);
#endif

#pragma unroll
  for (int t = 0; t < 4; ++t) {
    v8f cc = (t == 0) ? c0 : (t == 1) ? c1 : (t == 2) ? c2 : c3;
    const int gn = n0 + 16 * t + lrow;
    const float bv = g.bias ? g.bias[gn] : 0.0f;
#pragma unroll
    for (int r = 0; r < 8; ++r) {
      const int gm = m0 + khalf * 8 + r;
      float v = cc[r] + bv;
      if (EPI == 0) {
        g.outF[(size_t)gm * g.ldo + gn] = v;
      } else if (EPI == 1) {
        g.outF[(size_t)gm * g.ldo + gn] = selu_f(v);
      } else if (EPI == 2) {
        g.outB[(size_t)gm * g.ldo + gn] = f2bf(selu_f(v));
      } else {
        float xp = bf2f(g.resid[(size_t)gm * TH + gn]);
        g.outB[((size_t)gm * 2 + g.rowAdd) * TH + gn] = f2bf(xp - selu_f(v));
      }
    }
  }
}

// ---------------------------------------------------------------------------
// GRU gate fusion: h' = (1-z)*n + z*h ;  r=sig(ir+hr), z=sig(iz+hz),
// n=tanh(in + r*hn).  Writes bf16; dup=1 duplicates rows (tree expansion).
// ---------------------------------------------------------------------------
__global__ void gru_gate_k(const float* __restrict__ gi,
                           const float* __restrict__ gh,
                           const unsigned short* __restrict__ hOld,
                           unsigned short* __restrict__ hNew,
                           int dup, int M) {
  int i = blockIdx.x * blockDim.x + threadIdx.x;
  if (i >= M * TR) return;
  int m = i >> 8, j = i & (TR - 1);
  size_t o = (size_t)m * (3 * TR);
  float r = sigmoid_f(gi[o + j] + gh[o + j]);
  float z = sigmoid_f(gi[o + TR + j] + gh[o + TR + j]);
  float n = tanhf(gi[o + 2 * TR + j] + r * gh[o + 2 * TR + j]);
  float h = bf2f(hOld[(size_t)m * TR + j]);
  unsigned short hb16 = f2bf((1.0f - z) * n + z * h);
  if (dup) {
    hNew[((size_t)m * 2) * TR + j]     = hb16;
    hNew[((size_t)m * 2 + 1) * TR + j] = hb16;
  } else {
    hNew[(size_t)m * TR + j] = hb16;
  }
}

// ---------------------------------------------------------------------------
// hb = sigmoid([xc,hc] . W_hb + b); depth update. Wave(32) per row.
// Expansion: dOut[2m]=dOut[2m+1]=dIn[m]+hb.  Leaf: out[m*1025+1024]=dIn[m]+hb.
// ---------------------------------------------------------------------------
__global__ void hb_depth_k(const unsigned short* __restrict__ xc,
                           const unsigned short* __restrict__ hc, int ldh,
                           const float* __restrict__ Whb,
                           const float* __restrict__ bhb,
                           const float* __restrict__ dIn,
                           float* __restrict__ dOut,
                           float* __restrict__ outLeaf, int M) {
  const int lane = threadIdx.x & 31;
  const int wave = threadIdx.x >> 5;
  const int row  = blockIdx.x * (blockDim.x >> 5) + wave;
  if (row >= M) return;
  float acc = 0.0f;
  const unsigned short* xr = xc + (size_t)row * TH;
  for (int k = lane; k < TH; k += 32) acc += bf2f(xr[k]) * Whb[k];
  const unsigned short* hr = hc + (size_t)row * ldh;
  for (int k = lane; k < TR; k += 32) acc += bf2f(hr[k]) * Whb[TH + k];
#pragma unroll
  for (int off = 16; off > 0; off >>= 1) acc += __shfl_xor(acc, off, 32);
  if (lane == 0) {
    float d = dIn[row] + sigmoid_f(acc + bhb[0]);
    if (outLeaf) {
      outLeaf[(size_t)row * (TH + 1) + TH] = d;
    } else {
      dOut[(size_t)row * 2]     = d;
      dOut[(size_t)row * 2 + 1] = d;
    }
  }
}

// Copy leaf terminals: out[m, 0:H] = f32(xc_bf16[m, 0:H])  (ld out = H+1)
__global__ void copy_terminal_k(const unsigned short* __restrict__ xc,
                                float* __restrict__ out, int n) {
  int i = blockIdx.x * blockDim.x + threadIdx.x;
  if (i >= n) return;
  int m = i >> 10, h = i & (TH - 1);
  out[(size_t)m * (TH + 1) + h] = bf2f(xc[i]);
}

// ---------------------------------------------------------------------------
// Host orchestration
// ---------------------------------------------------------------------------
static inline int cdiv(int a, int b) { return (a + b - 1) / b; }

extern "C" void kernel_launch(void* const* d_in, const int* in_sizes, int n_in,
                              void* d_out, int out_size, void* d_ws, size_t ws_size,
                              hipStream_t stream) {
  (void)in_sizes; (void)n_in; (void)out_size; (void)ws_size;
  const float* x    = (const float*)d_in[0];   // [64,32,1024]
  const float* Whid = (const float*)d_in[1];   // [1024,1024]
  const float* bhid = (const float*)d_in[2];   // [1024]
  const float* Whb  = (const float*)d_in[3];   // [1,1280]
  const float* bhb  = (const float*)d_in[4];   // [1]
  const float* Wbr  = (const float*)d_in[5];   // [2048,1280]
  const float* bbr  = (const float*)d_in[6];   // [2048]
  const float* Wih  = (const float*)d_in[7];   // [768,1024]
  const float* Whh  = (const float*)d_in[8];   // [768,256]
  const float* bih  = (const float*)d_in[9];   // [768]
  const float* bhh  = (const float*)d_in[10];  // [768]
  float* out = (float*)d_out;                  // [64,512,1025]

  // ---- carve workspace (~450 MB) ----
  char* p = (char*)d_ws;
  auto carve = [&](size_t bytes) {
    char* r = p; p += (bytes + 255) & ~(size_t)255; return r;
  };
  const int MLEAF = TB << TDEPTH;  // 32768
  unsigned short* WhidB = (unsigned short*)carve((size_t)TH * TH * 2);
  unsigned short* WihB  = (unsigned short*)carve((size_t)3 * TR * TH * 2);
  unsigned short* WhhB  = (unsigned short*)carve((size_t)3 * TR * TR * 2);
  unsigned short* WbrB  = (unsigned short*)carve((size_t)2 * TH * (TH + TR) * 2);
  unsigned short* sxB   = (unsigned short*)carve((size_t)TB * TS * TH * 2);
  float*          yroot = (float*)carve((size_t)TB * TS * TH * 4);
  unsigned short* xcA   = (unsigned short*)carve((size_t)MLEAF * TH * 2);
  unsigned short* xcB   = (unsigned short*)carve((size_t)MLEAF * TH * 2);
  unsigned short* hcA   = (unsigned short*)carve((size_t)MLEAF * TR * 2);
  unsigned short* hcB   = (unsigned short*)carve((size_t)MLEAF * TR * 2);
  float*          dA    = (float*)carve((size_t)MLEAF * 4);
  float*          dB    = (float*)carve((size_t)MLEAF * 4);
  float*          gi    = (float*)carve((size_t)MLEAF * 3 * TR * 4);
  float*          gh    = (float*)carve((size_t)MLEAF * 3 * TR * 4);
  unsigned short* lrB   = (unsigned short*)carve((size_t)(MLEAF / 2) * 2 * TH * 2);

  const int T = 256;
  // ---- weight / input casts to bf16 ----
  cast_bf16_k<<<cdiv(TH * TH, T), T, 0, stream>>>(Whid, WhidB, TH * TH);
  cast_bf16_k<<<cdiv(3 * TR * TH, T), T, 0, stream>>>(Wih, WihB, 3 * TR * TH);
  cast_bf16_k<<<cdiv(3 * TR * TR, T), T, 0, stream>>>(Whh, WhhB, 3 * TR * TR);
  cast_bf16_k<<<cdiv(2 * TH * (TH + TR), T), T, 0, stream>>>(Wbr, WbrB, 2 * TH * (TH + TR));
  cast_selu_bf16_k<<<cdiv(TB * TS * TH, T), T, 0, stream>>>(x, sxB, TB * TS * TH);

  // ---- root: y = selu(selu(x) @ Whid^T + b); xc0 = mean_s(y) ----
  {
    GemmArgs a{};
    a.s0 = {sxB, TH, TH, WhidB, TH};
    a.bias = bhid; a.outF = yroot; a.ldo = TH;
    gemm_bf16_wmma<1><<<dim3(TH / 128, (TB * TS) / 64), 256, 0, stream>>>(a);
  }
  mean_rows_k<<<cdiv(TB * TH, T), T, 0, stream>>>(yroot, xcA);
  (void)hipMemsetAsync(hcA, 0, (size_t)TB * TR * 2, stream);
  (void)hipMemsetAsync(dA, 0, (size_t)TB * 4, stream);

  // ---- level-synchronous tree expansion ----
  for (int d = 0; d < TDEPTH; ++d) {
    const int M = TB << d;
    {  // gi = xc @ Wih^T + b_ih    [M, 768]
      GemmArgs a{};
      a.s0 = {xcA, TH, TH, WihB, TH};
      a.bias = bih; a.outF = gi; a.ldo = 3 * TR;
      gemm_bf16_wmma<0><<<dim3((3 * TR) / 128, M / 64), 256, 0, stream>>>(a);
    }
    {  // gh = hc @ Whh^T + b_hh    [M, 768]
      GemmArgs a{};
      a.s0 = {hcA, TR, TR, WhhB, TR};
      a.bias = bhh; a.outF = gh; a.ldo = 3 * TR;
      gemm_bf16_wmma<0><<<dim3((3 * TR) / 128, M / 64), 256, 0, stream>>>(a);
    }
    gru_gate_k<<<M, 256, 0, stream>>>(gi, gh, hcA, hcB, 1, M);
    hb_depth_k<<<M / 8, 256, 0, stream>>>(xcA, hcB, 2 * TR, Whb, bhb, dA, dB,
                                          nullptr, M);
    {  // lr = selu([xc, hc_new] @ Wbr^T + b_br) -> bf16 [M, 2048]
      GemmArgs a{};
      a.s0 = {xcA, TH, TH, WbrB, TH + TR};
      a.s1 = {hcB, 2 * TR, TR, WbrB + TH, TH + TR};  // dup rows -> lda 512
      a.bias = bbr; a.outB = lrB; a.ldo = 2 * TH;
      gemm_bf16_wmma<2><<<dim3((2 * TH) / 128, M / 64), 256, 0, stream>>>(a);
    }
    for (int c = 0; c < 2; ++c) {  // child = xc - selu(lr_half @ Whid^T + b)
      GemmArgs a{};
      a.s0 = {lrB + c * TH, 2 * TH, TH, WhidB, TH};
      a.bias = bhid; a.outB = xcB; a.resid = xcA; a.rowAdd = c; a.ldo = TH;
      gemm_bf16_wmma<3><<<dim3(TH / 128, M / 64), 256, 0, stream>>>(a);
    }
    // swap ping-pong
    unsigned short* t;
    t = xcA; xcA = xcB; xcB = t;
    t = hcA; hcA = hcB; hcB = t;
    float* tf = dA; dA = dB; dB = tf;
  }

  // ---- leaf step: final GRU + hb, assemble output ----
  {
    const int M = MLEAF;
    GemmArgs a{};
    a.s0 = {xcA, TH, TH, WihB, TH};
    a.bias = bih; a.outF = gi; a.ldo = 3 * TR;
    gemm_bf16_wmma<0><<<dim3((3 * TR) / 128, M / 64), 256, 0, stream>>>(a);
    GemmArgs b{};
    b.s0 = {hcA, TR, TR, WhhB, TR};
    b.bias = bhh; b.outF = gh; b.ldo = 3 * TR;
    gemm_bf16_wmma<0><<<dim3((3 * TR) / 128, M / 64), 256, 0, stream>>>(b);
    gru_gate_k<<<M, 256, 0, stream>>>(gi, gh, hcA, hcB, 0, M);
    hb_depth_k<<<M / 8, 256, 0, stream>>>(xcA, hcB, TR, Whb, bhb, dA, nullptr,
                                          out, M);
    copy_terminal_k<<<cdiv(M * TH, T), T, 0, stream>>>(xcA, out, M * TH);
  }
}
